// SimpleTransformerBlock_79448305041430
// MI455X (gfx1250) — matrix operand order repaired and checked by reading, not compiled, not run on hardware
//
#include <hip/hip_runtime.h>
#include <hip/hip_bf16.h>

// ---------------------------------------------------------------------------
// Problem constants (from reference): B=2, L=2048, D=1024, H=16, hd=64, F=4096
// ---------------------------------------------------------------------------
#define BB 2
#define LL 2048
#define DD 1024
#define FF 4096
#define HH 16
#define HD 64
#define BL (BB*LL)   // 4096 rows

typedef __attribute__((ext_vector_type(16))) __bf16 v16bf;
typedef __attribute__((ext_vector_type(8)))  float  v8f;
typedef int v4i_ __attribute__((vector_size(16)));

union Frag32B { uint4 q[2]; v16bf v; };   // 32 bytes = one WMMA bf16 A/B fragment

#if __has_builtin(__builtin_amdgcn_global_load_async_to_lds_b128)
#define ASYNC_LDS 1
#else
#define ASYNC_LDS 0
#endif

#if __has_builtin(__builtin_amdgcn_update_dpp)
#define HAVE_DPP 1
#else
#define HAVE_DPP 0
#endif

__device__ __forceinline__ v8f wmma_bf16(v16bf a, v16bf b, v8f c) {
  // D = A(16x32) * B(32x16) + C, f32 accumulate
  return __builtin_amdgcn_wmma_f32_16x16x32_bf16(false, a, false, b, (short)0, c,
                                                 false, false);
}

__device__ __forceinline__ float gelu_tanh(float x) {
  return 0.5f * x * (1.0f + tanhf(0.7978845608028654f * (x + 0.044715f * x * x * x)));
}

// VALU-only max reduction across a 16-lane row (DPP row_ror), shfl fallback.
#if HAVE_DPP
template <int N>
__device__ __forceinline__ float row_ror_f32(float x) {
  int r = __builtin_amdgcn_update_dpp(0, __float_as_int(x), 0x120 | N, 0xf, 0xf, true);
  return __int_as_float(r);
}
#endif
__device__ __forceinline__ float red16_max(float x) {
#if HAVE_DPP
  x = fmaxf(x, row_ror_f32<8>(x));
  x = fmaxf(x, row_ror_f32<4>(x));
  x = fmaxf(x, row_ror_f32<2>(x));
  x = fmaxf(x, row_ror_f32<1>(x));
#else
#pragma unroll
  for (int m = 8; m >= 1; m >>= 1) x = fmaxf(x, __shfl_xor(x, m, 32));
#endif
  return x;
}

// Async (ASYNCcnt-tracked) 32-byte global->LDS copy; falls back to VGPR path.
__device__ __forceinline__ void cp32_g2l(const __bf16* g, __bf16* l) {
#if ASYNC_LDS
  // builtin prototype: (int4 AS1*, int4 AS3*, Imm offset, Imm cpol);
  // the imm offset applies to BOTH global and LDS addresses.
  __attribute__((address_space(1))) v4i_* gp =
      (__attribute__((address_space(1))) v4i_*)(
          (__attribute__((address_space(1))) const void*)g);
  __attribute__((address_space(3))) v4i_* lp =
      (__attribute__((address_space(3))) v4i_*)(
          (__attribute__((address_space(3))) void*)l);
  __builtin_amdgcn_global_load_async_to_lds_b128(gp, lp, 0, 0);
  __builtin_amdgcn_global_load_async_to_lds_b128(gp, lp, 16, 0);
#else
  ((uint4*)l)[0] = ((const uint4*)g)[0];
  ((uint4*)l)[1] = ((const uint4*)g)[1];
#endif
}

__device__ __forceinline__ void wait_async0() {
#if ASYNC_LDS
#if __has_builtin(__builtin_amdgcn_s_wait_asynccnt)
  __builtin_amdgcn_s_wait_asynccnt(0);
#else
  asm volatile("s_wait_asynccnt 0x0" ::: "memory");
#endif
#endif
}

// ---------------------------------------------------------------------------
// f32 -> bf16 conversion (weights), vectorized x4
// ---------------------------------------------------------------------------
__global__ __launch_bounds__(256) void cvt_kernel(const float* __restrict__ in,
                                                  __bf16* __restrict__ out, int n) {
  int i = blockIdx.x * 256 + threadIdx.x;
  int idx = i * 4;
  if (idx < n) {
    float4 v = ((const float4*)in)[i];
    out[idx + 0] = (__bf16)v.x;
    out[idx + 1] = (__bf16)v.y;
    out[idx + 2] = (__bf16)v.z;
    out[idx + 3] = (__bf16)v.w;
  }
}

// ---------------------------------------------------------------------------
// LayerNorm over D=1024, one row per 256-thread block, bf16 output
// ---------------------------------------------------------------------------
__global__ __launch_bounds__(256) void ln_kernel(const float* __restrict__ x,
                                                 const float* __restrict__ sc,
                                                 const float* __restrict__ bs,
                                                 __bf16* __restrict__ out) {
  const int row = blockIdx.x;
  const int t = threadIdx.x;
  const int lane = t & 31, wv = t >> 5;
  float4 v = ((const float4*)(x + (size_t)row * DD))[t];
  float s  = v.x + v.y + v.z + v.w;
  float s2 = v.x * v.x + v.y * v.y + v.z * v.z + v.w * v.w;
#pragma unroll
  for (int m = 16; m >= 1; m >>= 1) {
    s  += __shfl_xor(s,  m, 32);
    s2 += __shfl_xor(s2, m, 32);
  }
  __shared__ float ps[8], ps2[8], stat[2];
  if (lane == 0) { ps[wv] = s; ps2[wv] = s2; }
  __syncthreads();
  if (t == 0) {
    float S = 0.f, S2 = 0.f;
#pragma unroll
    for (int i = 0; i < 8; i++) { S += ps[i]; S2 += ps2[i]; }
    float mu  = S * (1.0f / DD);
    float var = S2 * (1.0f / DD) - mu * mu;
    stat[0] = mu;
    stat[1] = rsqrtf(var + 1e-6f);
  }
  __syncthreads();
  const float mu = stat[0], rs = stat[1];
  const int c = t * 4;
  float4 sv = ((const float4*)sc)[t];
  float4 bv = ((const float4*)bs)[t];
  __bf16* o = out + (size_t)row * DD + c;
  o[0] = (__bf16)((v.x - mu) * rs * sv.x + bv.x);
  o[1] = (__bf16)((v.y - mu) * rs * sv.y + bv.y);
  o[2] = (__bf16)((v.z - mu) * rs * sv.z + bv.z);
  o[3] = (__bf16)((v.w - mu) * rs * sv.w + bv.w);
}

// ---------------------------------------------------------------------------
// Tiled WMMA GEMM: C[M,N] = A[M,K](bf16) * B[K,N](bf16) (+bias)(+resid)(gelu)
// Block tile 128x128, 8 waves (2x4), wave tile 64x32, K-step 32.
// Double-buffered LDS (2 x 16 KB) staged with global_load_async_to_lds_b128.
// EPI: 0 = scatter to q/kT/v buffers (q pre-scaled by 1/sqrt(hd)),
//      1 = f32 out = val+bias+resid, 2 = bf16 out = gelu(val+bias)
// ---------------------------------------------------------------------------
template <int EPI>
__global__ __launch_bounds__(256) void gemm_kernel(
    const __bf16* __restrict__ A, const __bf16* __restrict__ Bm,
    int M, int N, int K,
    const float* __restrict__ bias, const float* __restrict__ resid,
    float* __restrict__ outf, __bf16* __restrict__ outb,
    __bf16* __restrict__ qb, __bf16* __restrict__ kb, __bf16* __restrict__ vb) {
  __shared__ __align__(16) __bf16 As[2][128 * 32];
  __shared__ __align__(16) __bf16 Bs[2][32 * 128];
  const int t = threadIdx.x;
  const int lane = t & 31;
  const int wave = t >> 5;
  const int wm = wave >> 2;     // 0..1  (64 rows each)
  const int wn = wave & 3;      // 0..3  (32 cols each)
  const int hf = lane >> 4;     // lane half
  const int mcol = lane & 15;
  const int m0 = blockIdx.y * 128;
  const int n0 = blockIdx.x * 128;
  // per-thread staging coordinates (32 B each for A and B tile)
  const int arow = t >> 1, aseg = t & 1;   // A: 128 rows x 32 cols
  const int brow = t >> 3, bseg = t & 7;   // B: 32 rows x 128 cols

  v8f acc[4][2];
#pragma unroll
  for (int i = 0; i < 4; i++)
#pragma unroll
    for (int j = 0; j < 2; j++) acc[i][j] = (v8f){0.f,0.f,0.f,0.f,0.f,0.f,0.f,0.f};

  auto stage = [&](int kt, int buf) {
    cp32_g2l(A + (size_t)(m0 + arow) * K + kt * 32 + aseg * 16,
             &As[buf][arow * 32 + aseg * 16]);
    cp32_g2l(Bm + (size_t)(kt * 32 + brow) * N + n0 + bseg * 16,
             &Bs[buf][brow * 128 + bseg * 16]);
  };

  const int nk = K >> 5;
  stage(0, 0);
  for (int kt = 0; kt < nk; kt++) {
    wait_async0();
    __syncthreads();                 // buffer kt&1 fully staged, prev reads done
    if (kt + 1 < nk) stage(kt + 1, (kt + 1) & 1);
    const __bf16* as = As[kt & 1];
    const __bf16* bs = Bs[kt & 1];
    // B fragments: lane = N column (mcol), K = hf*8..+8 and 16+hf*8..+8
    Frag32B bf0, bf1;
    {
#pragma unroll
      for (int s = 0; s < 16; s++) {
        const int k = (s < 8) ? (hf * 8 + s) : (16 + hf * 8 + (s - 8));
        bf0.v[s] = bs[k * 128 + wn * 32 + mcol];
        bf1.v[s] = bs[k * 128 + wn * 32 + 16 + mcol];
      }
    }
#pragma unroll
    for (int mt = 0; mt < 4; mt++) {
      // A fragment: row = mcol, K = hf*8..+8 and 16+hf*8..+8
      Frag32B af;
      const __bf16* ap = as + (wm * 64 + mt * 16 + mcol) * 32;
      af.q[0] = *(const uint4*)(ap + hf * 8);
      af.q[1] = *(const uint4*)(ap + 16 + hf * 8);
      acc[mt][0] = wmma_bf16(af.v, bf0.v, acc[mt][0]);
      acc[mt][1] = wmma_bf16(af.v, bf1.v, acc[mt][1]);
    }
  }

  // Epilogue: C/D layout row = 8*hf + r, col = mcol
#pragma unroll
  for (int mt = 0; mt < 4; mt++)
#pragma unroll
    for (int nt = 0; nt < 2; nt++) {
#pragma unroll
      for (int r = 0; r < 8; r++) {
        int gm = m0 + wm * 64 + mt * 16 + hf * 8 + r;
        int gn = n0 + wn * 32 + nt * 16 + mcol;
        float val = acc[mt][nt][r];
        if (EPI == 0) {
          // qkv scatter: q [B,H,L,hd] (scaled), kT [B,H,hd,L], v [B,H,L,hd]
          int b = gm >> 11, l = gm & 2047;
          int part = gn >> 10, c = gn & 1023;
          int h = c >> 6, w = c & 63;
          if (part == 0)
            qb[((size_t)((b * HH + h) * LL + l)) * HD + w] = (__bf16)(val * 0.125f);
          else if (part == 1)
            kb[((size_t)((b * HH + h) * HD + w)) * LL + l] = (__bf16)val;
          else
            vb[((size_t)((b * HH + h) * LL + l)) * HD + w] = (__bf16)val;
        } else if (EPI == 1) {
          outf[(size_t)gm * N + gn] = val + bias[gn] + resid[(size_t)gm * N + gn];
        } else {
          outb[(size_t)gm * N + gn] = (__bf16)gelu_tanh(val + bias[gn]);
        }
      }
    }
}

// ---------------------------------------------------------------------------
// Flash attention, one 16-row q-tile per wave, 4 waves/block.
// q [B,H,L,hd] bf16 (pre-scaled by 1/8), kT [B,H,hd,L] bf16, v [B,H,L,hd] bf16
// Row max: DPP reduction (VALU only). Row sums: P @ ones via WMMA (no shuffles).
// ---------------------------------------------------------------------------
__global__ __launch_bounds__(128) void attn_kernel(const __bf16* __restrict__ q,
                                                   const __bf16* __restrict__ kT,
                                                   const __bf16* __restrict__ v,
                                                   __bf16* __restrict__ o) {
  const int wid = threadIdx.x >> 5;
  const int lane = threadIdx.x & 31;
  const int hf = lane >> 4;
  const int col = lane & 15;
  const int tile = blockIdx.x * 4 + wid;   // 0..4095
  const int bh = tile >> 7;                // / (2048/16)
  const int q0 = (tile & 127) * 16;
  const __bf16* qbase = q  + ((size_t)bh * LL + q0) * HD;
  const __bf16* kbase = kT + (size_t)bh * HD * LL;
  const __bf16* vbase = v  + (size_t)bh * LL * HD;

  __shared__ __align__(16) __bf16 Pb[4][16 * 64];
  __bf16* P = Pb[wid];

  // ones B-fragment for WMMA row sums
  v16bf ones;
#pragma unroll
  for (int e = 0; e < 16; e++) ones[e] = (__bf16)1.0f;

  // Q fragments (A layout): row = col (m), K split by lane half; hd=64 -> 2 frags
  Frag32B qa[2];
  {
    const __bf16* qrow = qbase + (size_t)col * HD;
#pragma unroll
    for (int kc = 0; kc < 2; kc++) {
      qa[kc].q[0] = *(const uint4*)(qrow + kc * 32 + hf * 8);
      qa[kc].q[1] = *(const uint4*)(qrow + kc * 32 + 16 + hf * 8);
    }
  }

  v8f accv[4];
#pragma unroll
  for (int i = 0; i < 4; i++) accv[i] = (v8f){0.f,0.f,0.f,0.f,0.f,0.f,0.f,0.f};
  float mrun[8], lsum[8];
#pragma unroll
  for (int r = 0; r < 8; r++) { mrun[r] = -1e30f; lsum[r] = 0.f; }

  for (int kb = 0; kb < LL; kb += 64) {
    // ---- scores: S[16 x 64] = Q * K^T, 8 WMMAs ----
    v8f sf[4];
#pragma unroll
    for (int i = 0; i < 4; i++) sf[i] = (v8f){0.f,0.f,0.f,0.f,0.f,0.f,0.f,0.f};
#pragma unroll
    for (int nt = 0; nt < 4; nt++) {
#pragma unroll
      for (int kc = 0; kc < 2; kc++) {
        Frag32B bf;  // lane = key column col, hd K in table order by lane half
#pragma unroll
        for (int s = 0; s < 16; s++) {
          const int k = (s < 8) ? (hf * 8 + s) : (16 + hf * 8 + (s - 8));
          bf.v[s] = kbase[(size_t)(kc * 32 + k) * LL + kb + nt * 16 + col];
        }
        sf[nt] = wmma_bf16(qa[kc].v, bf.v, sf[nt]);
      }
    }
    // ---- online softmax (rows = 8*hf + r; 16 lanes share a row) ----
    float corr[8];
#pragma unroll
    for (int r = 0; r < 8; r++) {
      float mx = fmaxf(fmaxf(sf[0][r], sf[1][r]), fmaxf(sf[2][r], sf[3][r]));
      mx = red16_max(mx);                       // VALU DPP reduction
      float mnew = fmaxf(mrun[r], mx);
      corr[r] = __expf(mrun[r] - mnew);
      mrun[r] = mnew;
    }
#pragma unroll
    for (int r = 0; r < 8; r++) {
      const int prow = hf * 8 + r;
#pragma unroll
      for (int nt = 0; nt < 4; nt++) {
        float p = __expf(sf[nt][r] - mrun[r]);
        P[prow * 64 + nt * 16 + col] = (__bf16)p;   // C-layout -> LDS
      }
    }
    // ---- P back in A layout from LDS ----
    Frag32B pa[2];
    {
      const __bf16* prow = P + (size_t)col * 64;
#pragma unroll
      for (int kc = 0; kc < 2; kc++) {
        pa[kc].q[0] = *(const uint4*)(prow + kc * 32 + hf * 8);
        pa[kc].q[1] = *(const uint4*)(prow + kc * 32 + 16 + hf * 8);
      }
    }
    // ---- row sums via WMMA: P @ ones broadcasts rowsum to every column ----
    v8f sums = (v8f){0.f,0.f,0.f,0.f,0.f,0.f,0.f,0.f};
    sums = wmma_bf16(pa[0].v, ones, sums);
    sums = wmma_bf16(pa[1].v, ones, sums);
#pragma unroll
    for (int r = 0; r < 8; r++) {
      lsum[r] = lsum[r] * corr[r] + sums[r];
#pragma unroll
      for (int nt2 = 0; nt2 < 4; nt2++) accv[nt2][r] = accv[nt2][r] * corr[r];
    }
    // ---- P @ V, 8 WMMAs ----
#pragma unroll
    for (int nt2 = 0; nt2 < 4; nt2++) {
#pragma unroll
      for (int kc = 0; kc < 2; kc++) {
        Frag32B vf;  // lane = hd column col, key K in table order by lane half
#pragma unroll
        for (int s = 0; s < 16; s++) {
          const int k = (s < 8) ? (hf * 8 + s) : (16 + hf * 8 + (s - 8));
          vf.v[s] = vbase[(size_t)(kb + kc * 32 + k) * HD + nt2 * 16 + col];
        }
        accv[nt2] = wmma_bf16(pa[kc].v, vf.v, accv[nt2]);
      }
    }
  }

  // ---- writeback: attn_out[b*L + l, h*64 + n] ----
  const int b = bh >> 4, h = bh & 15;
#pragma unroll
  for (int r = 0; r < 8; r++) {
    float inv = 1.0f / lsum[r];
    size_t row = (size_t)(b * LL + q0 + hf * 8 + r);
#pragma unroll
    for (int nt2 = 0; nt2 < 4; nt2++) {
      o[row * DD + h * HD + nt2 * 16 + col] = (__bf16)(accv[nt2][r] * inv);
    }
  }
}

// ---------------------------------------------------------------------------
// Orchestration
// ---------------------------------------------------------------------------
extern "C" void kernel_launch(void* const* d_in, const int* in_sizes, int n_in,
                              void* d_out, int out_size, void* d_ws, size_t ws_size,
                              hipStream_t stream) {
  const float* x      = (const float*)d_in[0];
  const float* ln1_s  = (const float*)d_in[1];
  const float* ln1_b  = (const float*)d_in[2];
  const float* w_qkv  = (const float*)d_in[3];
  const float* w_proj = (const float*)d_in[4];
  const float* b_proj = (const float*)d_in[5];
  const float* ln2_s  = (const float*)d_in[6];
  const float* ln2_b  = (const float*)d_in[7];
  const float* w_mlp1 = (const float*)d_in[8];
  const float* b_mlp1 = (const float*)d_in[9];
  const float* w_mlp2 = (const float*)d_in[10];
  const float* b_mlp2 = (const float*)d_in[11];
  float* out = (float*)d_out;
  (void)in_sizes; (void)n_in; (void)out_size; (void)ws_size;

  char* ws = (char*)d_ws;
  size_t off = 0;
  auto take = [&](size_t bytes) -> char* {
    char* p = ws + off;
    off += (bytes + 255) & ~(size_t)255;
    return p;
  };
  __bf16* hbf   = (__bf16*)take((size_t)BL * DD * 2);       // ln1 out (reused as h2)
  __bf16* wqkvb = (__bf16*)take((size_t)DD * 3 * DD * 2);
  __bf16* wprjb = (__bf16*)take((size_t)DD * DD * 2);
  __bf16* wm1b  = (__bf16*)take((size_t)DD * FF * 2);
  __bf16* wm2b  = (__bf16*)take((size_t)FF * DD * 2);
  __bf16* qb    = (__bf16*)take((size_t)BL * DD * 2);
  __bf16* kb    = (__bf16*)take((size_t)BL * DD * 2);       // K transposed
  __bf16* vb    = (__bf16*)take((size_t)BL * DD * 2);
  __bf16* aout  = (__bf16*)take((size_t)BL * DD * 2);
  float*  x1    = (float*)take((size_t)BL * DD * 4);
  __bf16* mbuf  = qb;    // 32 MB alias over q/kT/v/aout (dead after proj GEMM)
  __bf16* h2bf  = hbf;   // alias (h dead after qkv GEMM)

  // 1) weight conversions to bf16
  cvt_kernel<<<(DD * 3 * DD) / 1024, 256, 0, stream>>>(w_qkv, wqkvb, DD * 3 * DD);
  cvt_kernel<<<(DD * DD) / 1024, 256, 0, stream>>>(w_proj, wprjb, DD * DD);
  cvt_kernel<<<(DD * FF) / 1024, 256, 0, stream>>>(w_mlp1, wm1b, DD * FF);
  cvt_kernel<<<(FF * DD) / 1024, 256, 0, stream>>>(w_mlp2, wm2b, FF * DD);

  // 2) LN1
  ln_kernel<<<BL, 256, 0, stream>>>(x, ln1_s, ln1_b, hbf);

  // 3) qkv GEMM (scatter into q / kT / v; q pre-scaled by 1/8)
  gemm_kernel<0><<<dim3(3 * DD / 128, BL / 128), 256, 0, stream>>>(
      hbf, wqkvb, BL, 3 * DD, DD, nullptr, nullptr, nullptr, nullptr, qb, kb, vb);

  // 4) flash attention
  attn_kernel<<<(BB * HH * (LL / 16)) / 4, 128, 0, stream>>>(qb, kb, vb, aout);

  // 5) proj + bias + residual(x) -> x1 (f32)
  gemm_kernel<1><<<dim3(DD / 128, BL / 128), 256, 0, stream>>>(
      aout, wprjb, BL, DD, DD, b_proj, x, x1, nullptr, nullptr, nullptr, nullptr);

  // 6) LN2
  ln_kernel<<<BL, 256, 0, stream>>>(x1, ln2_s, ln2_b, h2bf);

  // 7) mlp1 + bias + gelu -> m (bf16)
  gemm_kernel<2><<<dim3(FF / 128, BL / 128), 256, 0, stream>>>(
      h2bf, wm1b, BL, FF, DD, b_mlp1, nullptr, nullptr, mbuf, nullptr, nullptr, nullptr);

  // 8) mlp2 + bias + residual(x1) -> out (f32)
  gemm_kernel<1><<<dim3(DD / 128, BL / 128), 256, 0, stream>>>(
      mbuf, wm2b, BL, DD, FF, b_mlp2, x1, out, nullptr, nullptr, nullptr, nullptr);
}
